// GATLayerWithOrder_1932735283942
// MI455X (gfx1250) — matrix operand order repaired and checked by reading, not compiled, not run on hardware
//
#include <hip/hip_runtime.h>
#include <hip/hip_bf16.h>

// ---------------------------------------------------------------------------
// GATLayerWithOrder for MI455X (gfx1250, wave32, WMMA).
// B=64, N=128, F_IN=F_OUT=256, E_DIM=64, CI=128.
// ---------------------------------------------------------------------------

typedef __attribute__((ext_vector_type(16))) _Float16 v16h;
typedef __attribute__((ext_vector_type(8)))  float    v8f;

__device__ __forceinline__ v8f wmma_f16(v16h a, v16h b, v8f c) {
  // D = A(16x32 f16) * B(32x16 f16) + C(16x16 f32)
  return __builtin_amdgcn_wmma_f32_16x16x32_f16(
      /*neg_a=*/false, a, /*neg_b=*/false, b,
      /*c_mod=*/(short)0, c, /*reuse_a=*/false, /*reuse_b=*/false);
}

__device__ __forceinline__ float lrelu(float x) { return x > 0.f ? x : 0.01f * x; }
__device__ __forceinline__ float sigm(float x)  { return 1.f / (1.f + __expf(-x)); }

// ---------------------------------------------------------------------------
// Generic WMMA GEMM: C[M,N] = A[M,K] * B    (f32 in/out, f16 WMMA operands)
//   BT=false: B is (K,N) row-major, ld=N.  BT=true: B is (N,K) row-major, ld=K.
// Block = 256 threads = 8 waves, each wave owns one 16x16 tile -> 32x64 / block.
// All launches use M%32==0, N%64==0, K%32==0 (true for every GEMM here).
// Batched via blockIdx.z with element strides sA/sB/sC.
// ---------------------------------------------------------------------------
template <bool BT>
__global__ void gemm_wmma(const float* __restrict__ A, const float* __restrict__ B,
                          float* __restrict__ C, int M, int N, int K,
                          long sA, long sB, long sC) {
  const float* Ab = A + (long)blockIdx.z * sA;
  const float* Bb = B + (long)blockIdx.z * sB;
  float*       Cb = C + (long)blockIdx.z * sC;

  const int wave = threadIdx.x >> 5;
  const int lane = threadIdx.x & 31;
  const int lh   = lane & 15;
  const int hi   = (lane >> 4) & 1;

  const int m0 = blockIdx.y * 32 + (wave >> 2) * 16;
  const int n0 = blockIdx.x * 64 + (wave & 3) * 16;

  v8f acc = {};
  const float* arow = Ab + (long)(m0 + lh) * K;

  for (int kc = 0; kc < K; kc += 32) {
    __builtin_prefetch(arow + kc + 64, 0, 1);  // global_prefetch_b8
    v16h a, b;
#pragma unroll
    for (int e = 0; e < 16; ++e) {
      // A 16x32, contiguous k order used here (same as B): K = e + (lane>=16 ? 16 : 0)
      int ka = kc + e + hi * 16;
      a[e] = (_Float16)arow[ka];
    }
#pragma unroll
    for (int e = 0; e < 16; ++e) {
      // B 32x16, same contiguous k order as A: N = lane&15, K = e + (lane>=16 ? 16 : 0)
      int kb = kc + e + hi * 16;
      float v = BT ? Bb[(long)(n0 + lh) * K + kb]
                   : Bb[(long)kb * N + (n0 + lh)];
      b[e] = (_Float16)v;
    }
    acc = wmma_f16(a, b, acc);
  }
#pragma unroll
  for (int j = 0; j < 8; ++j) {
    int row = m0 + j + hi * 8;          // f32 C/D layout
    Cb[(long)row * N + (n0 + lh)] = acc[j];
  }
}

// ---------------------------------------------------------------------------
// s[row] = dot(af[row, 0:256], fcc_w)   — one 64-thread block per row
// ---------------------------------------------------------------------------
__global__ void dot_rows(const float* __restrict__ A, const float* __restrict__ w,
                         float* __restrict__ out, int K) {
  __shared__ float red[64];
  const long row = blockIdx.x;
  float p = 0.f;
  for (int k = threadIdx.x; k < K; k += 64) p += A[row * K + k] * w[k];
  red[threadIdx.x] = p;
  __syncthreads();
  for (int s = 32; s > 0; s >>= 1) {
    if (threadIdx.x < (unsigned)s) red[threadIdx.x] += red[threadIdx.x + s];
    __syncthreads();
  }
  if (threadIdx.x == 0) out[row] = red[0];
}

// ---------------------------------------------------------------------------
// Attention scores + mask + softmax + hn, one (b,i) row per 128-thread block.
//   e[b,i,j] = fc_b + sum_d lrelu(f1[b,p,d] + f2[b,q,d]) * fc_w[d]
//   i<64 : p = q = 2i + (j>=64)
//   i>=64: p = (2j)%128, q = (2j+1)%128
//   att = softmax_j(where(adj>0, e, -9e15));  hn = lrelu(att*s[b,i] + fcc_b)
// ---------------------------------------------------------------------------
__global__ void att_softmax_hn(const float* __restrict__ f1, const float* __restrict__ f2,
                               const int* __restrict__ adj, const float* __restrict__ fc_w,
                               const float* __restrict__ fc_b_p,
                               const float* __restrict__ s, const float* __restrict__ fcc_b_p,
                               float* __restrict__ att, float* __restrict__ hn) {
  const int bi = blockIdx.x;          // b*128 + i
  const int b = bi >> 7, i = bi & 127;
  const int j = threadIdx.x;          // 0..127
  int p, q;
  if (i < 64) { p = 2 * i + (j >= 64 ? 1 : 0); q = p; }
  else        { p = (2 * j) & 127;    q = (2 * j + 1) & 127; }

  const float* F1 = f1 + ((long)b * 128 + p) * 64;
  const float* F2 = f2 + ((long)b * 128 + q) * 64;
  float e = 0.f;
#pragma unroll 8
  for (int d = 0; d < 64; ++d) e += lrelu(F1[d] + F2[d]) * fc_w[d];
  e += fc_b_p[0];

  const long rowo = ((long)b * 128 + i) * 128;
  const float masked = (adj[rowo + j] > 0) ? e : -9.0e15f;

  __shared__ float red[128];
  red[j] = masked; __syncthreads();
  for (int st = 64; st > 0; st >>= 1) {
    if (j < st) red[j] = fmaxf(red[j], red[j + st]);
    __syncthreads();
  }
  const float mx = red[0]; __syncthreads();
  const float ex = __expf(masked - mx);
  red[j] = ex; __syncthreads();
  for (int st = 64; st > 0; st >>= 1) {
    if (j < st) red[j] += red[j + st];
    __syncthreads();
  }
  const float av = ex / red[0];
  att[rowo + j] = av;
  hn[rowo + j]  = lrelu(av * s[(long)b * 128 + i] + fcc_b_p[0]);
}

// ---------------------------------------------------------------------------
// f32 -> f16 conversion (whh weights, consumed every LSTM step)
// ---------------------------------------------------------------------------
__global__ void cvt_f16(const float* __restrict__ in, _Float16* __restrict__ out, long n) {
  long i = (long)blockIdx.x * blockDim.x + threadIdx.x;
  if (i < n) out[i] = (_Float16)in[i];
}

// ---------------------------------------------------------------------------
// Persistent bidirectional LSTM scan. grid = 2 blocks (dir), 256 threads.
// gx = hn @ wih.T precomputed (rows = b*128+t, cols = 512 gates i|f|g|o).
// h (64x128) kept in LDS as f16; c kept in registers: wave w owns batch-tile
// m = w>>1 and hidden-tiles u = 4*(w&1)+q (q=0..3), and for each patch also
// owns all 4 gate tiles -> gate math fully in registers, no g staging buffer.
// Per step: 8 waves x 64 v_wmma, two barriers.
// ---------------------------------------------------------------------------
__global__ __launch_bounds__(256, 1)
void lstm_scan(const float* __restrict__ gx_f, const float* __restrict__ gx_r,
               const _Float16* __restrict__ whh_f16, const _Float16* __restrict__ whh_r16,
               const float* __restrict__ b_f, const float* __restrict__ b_r,
               float* __restrict__ hf, float* __restrict__ hr) {
  __shared__ _Float16 ldsH[64 * 128];

  const int dir = blockIdx.x;
  const float*    gx   = dir ? gx_r   : gx_f;
  const _Float16* whh  = dir ? whh_r16 : whh_f16;
  const float*    bias = dir ? b_r    : b_f;
  float*          hout = dir ? hr     : hf;

  const int wave  = threadIdx.x >> 5;
  const int lane  = threadIdx.x & 31;
  const int lh    = lane & 15;
  const int hi    = (lane >> 4) & 1;
  const int m     = wave >> 1;            // batch tile 0..3   (rows m*16..)
  const int ubase = (wave & 1) * 4;       // hidden tiles ubase..ubase+3

  for (int t = threadIdx.x; t < 64 * 128; t += 256) ldsH[t] = (_Float16)0.f;
  __syncthreads();

  v8f c[4] = {};                          // cell state, register-resident

  for (int step = 0; step < 128; ++step) {
    const int t = dir ? (127 - step) : step;

    v8f acc[4][4] = {};                   // [gate][patch]
    for (int kc = 0; kc < 128; kc += 32) {
      v16h a;
#pragma unroll
      for (int e = 0; e < 16; ++e) {
        int ka = kc + e + hi * 16;
        a[e] = ldsH[(m * 16 + lh) * 128 + ka];
      }
#pragma unroll
      for (int q = 0; q < 4; ++q) {
        const int u = ubase + q;
#pragma unroll
        for (int gt = 0; gt < 4; ++gt) {
          v16h bf;
          const int nrow = gt * 128 + u * 16 + lh;   // whh row (B^T layout)
#pragma unroll
          for (int e = 0; e < 16; ++e)
            bf[e] = whh[(long)nrow * 128 + kc + e + hi * 16];
          acc[gt][q] = wmma_f16(a, bf, acc[gt][q]);
        }
      }
    }
    __syncthreads();                      // all reads of ldsH done

#pragma unroll
    for (int q = 0; q < 4; ++q) {
      const int u = ubase + q;
#pragma unroll
      for (int j = 0; j < 8; ++j) {
        const int brow = m * 16 + j + hi * 8;     // batch index 0..63
        const int ucol = u * 16 + lh;             // hidden index 0..127
        const long gbase = ((long)brow * 128 + t) * 512;
        const float gi = acc[0][q][j] + gx[gbase +       ucol] + bias[      ucol];
        const float gf = acc[1][q][j] + gx[gbase + 128 + ucol] + bias[128 + ucol];
        const float gg = acc[2][q][j] + gx[gbase + 256 + ucol] + bias[256 + ucol];
        const float go = acc[3][q][j] + gx[gbase + 384 + ucol] + bias[384 + ucol];
        const float cn = sigm(gf) * c[q][j] + sigm(gi) * tanhf(gg);
        c[q][j] = cn;
        const float h = sigm(go) * tanhf(cn);
        ldsH[brow * 128 + ucol] = (_Float16)h;
        hout[((long)brow * 128 + t) * 128 + ucol] = h;
      }
    }
    __syncthreads();                      // new h visible to all waves
  }
}

// ---------------------------------------------------------------------------
// Final fused GEMM: out = elu( concat(lrelu(hf), lrelu(hr), h1) @ Wo + bo )
// M=8192, K=512, N=256. Concat/lrelu fused into A-operand load, bias+ELU
// fused into the epilogue.
// ---------------------------------------------------------------------------
__global__ void out_gemm(const float* __restrict__ hf, const float* __restrict__ hr,
                         const float* __restrict__ h1, const float* __restrict__ Wo,
                         const float* __restrict__ bo, float* __restrict__ out) {
  const int wave = threadIdx.x >> 5;
  const int lane = threadIdx.x & 31;
  const int lh   = lane & 15;
  const int hi   = (lane >> 4) & 1;
  const int m0 = blockIdx.y * 32 + (wave >> 2) * 16;
  const int n0 = blockIdx.x * 64 + (wave & 3) * 16;
  const int row = m0 + lh;

  v8f acc = {};
  for (int kc = 0; kc < 512; kc += 32) {
    v16h a, b;
#pragma unroll
    for (int e = 0; e < 16; ++e) {
      int k = kc + e + hi * 16;
      float v;
      if (k < 128)      v = lrelu(hf[(long)row * 128 + k]);
      else if (k < 256) v = lrelu(hr[(long)row * 128 + (k - 128)]);
      else              v = h1[(long)row * 256 + (k - 256)];
      a[e] = (_Float16)v;
    }
#pragma unroll
    for (int e = 0; e < 16; ++e) {
      int k = kc + e + hi * 16;
      b[e] = (_Float16)Wo[(long)k * 256 + (n0 + lh)];
    }
    acc = wmma_f16(a, b, acc);
  }
#pragma unroll
  for (int j = 0; j < 8; ++j) {
    const int r = m0 + j + hi * 8;
    const int cidx = n0 + lh;
    const float v = acc[j] + bo[cidx];
    out[(long)r * 256 + cidx] = v > 0.f ? v : (__expf(v) - 1.f);
  }
}

// ---------------------------------------------------------------------------
extern "C" void kernel_launch(void* const* d_in, const int* in_sizes, int n_in,
                              void* d_out, int out_size, void* d_ws, size_t ws_size,
                              hipStream_t stream) {
  constexpr int B = 64, N = 128, F = 256, E = 64, CI = 128;
  constexpr long R = (long)B * N;          // 8192 rows

  const float* input   = (const float*)d_in[0];
  const int*   adj     = (const int*)  d_in[1];
  const float* W       = (const float*)d_in[2];
  const float* a       = (const float*)d_in[3];
  const float* fc_w    = (const float*)d_in[4];
  const float* fc_b    = (const float*)d_in[5];
  const float* fcc_w   = (const float*)d_in[6];
  const float* fcc_b   = (const float*)d_in[7];
  const float* fc_o_w  = (const float*)d_in[8];
  const float* fc_o_b  = (const float*)d_in[9];
  const float* wih_f   = (const float*)d_in[10];
  const float* whh_f   = (const float*)d_in[11];
  const float* b_f     = (const float*)d_in[12];
  const float* wih_r   = (const float*)d_in[13];
  const float* whh_r   = (const float*)d_in[14];
  const float* b_r     = (const float*)d_in[15];

  // workspace layout (floats)
  float* ws  = (float*)d_ws;
  float* af  = ws;               // R*F      = 2,097,152
  float* f1  = af  + R * F;      // R*E      =   524,288
  float* f2  = f1  + R * E;      // R*E
  float* sbf = f2  + R * E;      // R
  float* att = sbf + R;          // R*N      = 1,048,576
  float* hn  = att + R * N;      // R*N
  float* gxf = hn  + R * N;      // R*4CI    = 4,194,304
  float* gxr = gxf + R * 4 * CI; // R*4CI
  float* h1  = gxr + R * 4 * CI; // R*F      = 2,097,152
  float* hfb = h1  + R * F;      // R*CI     = 1,048,576
  float* hrb = hfb + R * CI;     // R*CI
  _Float16* whhf16 = (_Float16*)(hrb + R * CI);   // 4CI*CI = 65,536 halfs
  _Float16* whhr16 = whhf16 + 4 * CI * CI;

  // 1) af = input @ W            (8192x256 @ 256x256)
  gemm_wmma<false><<<dim3(F / 64, R / 32, 1), 256, 0, stream>>>(
      input, W, af, (int)R, F, F, 0, 0, 0);

  // 2) f1 = af @ a[:256], f2 = af @ a[256:]   (8192x256 @ 256x64)
  gemm_wmma<false><<<dim3(1, R / 32, 1), 256, 0, stream>>>(
      af, a,         f1, (int)R, E, F, 0, 0, 0);
  gemm_wmma<false><<<dim3(1, R / 32, 1), 256, 0, stream>>>(
      af, a + F * E, f2, (int)R, E, F, 0, 0, 0);

  // 3) s = af @ fcc_w
  dot_rows<<<(int)R, 64, 0, stream>>>(af, fcc_w, sbf, F);

  // 4) scores + masked softmax + hn (fused)
  att_softmax_hn<<<(int)R, N, 0, stream>>>(f1, f2, adj, fc_w, fc_b, sbf, fcc_b,
                                           att, hn);

  // 5) h1 = att @ af  (batched over B: 128x128 @ 128x256)
  gemm_wmma<false><<<dim3(F / 64, N / 32, B), 256, 0, stream>>>(
      att, af, h1, N, F, N, (long)N * N, (long)N * F, (long)N * F);

  // 6) gx = hn @ wih.T + (bias added in scan)   (8192x128 @ 128x512)
  gemm_wmma<true><<<dim3(4 * CI / 64, R / 32, 1), 256, 0, stream>>>(
      hn, wih_f, gxf, (int)R, 4 * CI, CI, 0, 0, 0);
  gemm_wmma<true><<<dim3(4 * CI / 64, R / 32, 1), 256, 0, stream>>>(
      hn, wih_r, gxr, (int)R, 4 * CI, CI, 0, 0, 0);

  // 7) recurrent weights -> f16 (reused every step, L2-resident)
  cvt_f16<<<(4 * CI * CI + 255) / 256, 256, 0, stream>>>(whh_f, whhf16, 4 * CI * CI);
  cvt_f16<<<(4 * CI * CI + 255) / 256, 256, 0, stream>>>(whh_r, whhr16, 4 * CI * CI);

  // 8) persistent bidirectional scan: one workgroup per direction
  lstm_scan<<<2, 256, 0, stream>>>(gxf, gxr, whhf16, whhr16, b_f, b_r, hfb, hrb);

  // 9) out = elu(concat(lrelu(hf), lrelu(hr), h1) @ fc_out_w + fc_out_b)
  out_gemm<<<dim3(F / 64, R / 32, 1), 256, 0, stream>>>(
      hfb, hrb, h1, fc_o_w, fc_o_b, (float*)d_out);
}
